// CamembertLayer_70016556859435
// MI455X (gfx1250) — matrix operand order repaired and checked by reading, not compiled, not run on hardware
//
#include <hip/hip_runtime.h>
#include <math.h>

// ---------------------------------------------------------------------------
// Types
// ---------------------------------------------------------------------------
typedef __bf16 bf16_t;
typedef __attribute__((ext_vector_type(8)))  __bf16 v8bf;
typedef __attribute__((ext_vector_type(16))) __bf16 v16bf;
typedef __attribute__((ext_vector_type(8)))  float  v8f;

#define HDIM 1024
#define SEQ  2048
#define BATCH 4
#define NHEADS 16
#define HEADD 64
#define FFDIM 4096

// ---------------------------------------------------------------------------
// WMMA helpers (CDNA5: D = A(16x32 bf16) * B(32x16 bf16) + C(16x16 f32))
// ---------------------------------------------------------------------------
static __device__ __forceinline__ v8f wmma_bf16(v16bf a, v16bf b, v8f c) {
    return __builtin_amdgcn_wmma_f32_16x16x32_bf16(
        /*neg_a=*/false, a, /*neg_b=*/false, b,
        /*c_mod=*/(short)0, c, /*reuse_a=*/false, /*reuse_b=*/false);
}

// A fragment: 16x32 bf16. lane<16 holds row=lane, K {0..7,16..23};
// lane>=16 holds row=lane-16, K {8..15,24..31}.  (ISA 7.12.2)
static __device__ __forceinline__ v16bf load_a_frag(const bf16_t* base, int ld) {
    const int lane = threadIdx.x & 31;
    const bf16_t* p = base + (lane & 15) * ld + ((lane >> 4) << 3);
    v8bf lo = *(const v8bf*)p;          // K = kh .. kh+7
    v8bf hi = *(const v8bf*)(p + 16);   // K = kh+16 .. kh+23
    return __builtin_shufflevector(lo, hi, 0,1,2,3,4,5,6,7,8,9,10,11,12,13,14,15);
}

// B fragment: 32x16 bf16, stored "Bt" = [N][K] (column-major of B).
// lane<16 holds column=lane, K {0..7,16..23}; lane>=16 column=lane-16, K {8..15,24..31}.
static __device__ __forceinline__ v16bf load_b_frag(const bf16_t* base, int ld) {
    const int lane = threadIdx.x & 31;
    const bf16_t* p = base + (lane & 15) * ld + ((lane >> 4) << 3);
    v8bf lo = *(const v8bf*)p;
    v8bf hi = *(const v8bf*)(p + 16);
    return __builtin_shufflevector(lo, hi, 0,1,2,3,4,5,6,7,8,9,10,11,12,13,14,15);
}

// Wave-relative LDS byte offset: generic LDS pointer low 32 bits ARE the
// LDS offset (ISA 10.2).
static __device__ __forceinline__ uint32_t lds_off_u32(const void* p) {
    return (uint32_t)(uintptr_t)p;
}

// 64B async copy global->LDS (4 x b128, ASYNCcnt-tracked, no data VGPRs).
// INST_OFFSET applies to BOTH lds and global address (ISA 08 section 4.4).
static __device__ __forceinline__ void async_copy64(uint32_t lds, uint64_t gaddr) {
    asm volatile(
        "global_load_async_to_lds_b128 %0, %1, off\n\t"
        "global_load_async_to_lds_b128 %0, %1, off offset:16\n\t"
        "global_load_async_to_lds_b128 %0, %1, off offset:32\n\t"
        "global_load_async_to_lds_b128 %0, %1, off offset:48"
        :: "v"(lds), "v"(gaddr) : "memory");
}

static __device__ __forceinline__ void wait_async0() {
    asm volatile("s_wait_asynccnt 0" ::: "memory");
}

// ---------------------------------------------------------------------------
// fp32 -> bf16 convert (elementwise)
// ---------------------------------------------------------------------------
__global__ __launch_bounds__(256) void cvt_kernel(const float* __restrict__ x,
                                                  bf16_t* __restrict__ y, int n) {
    int i = (blockIdx.x * 256 + threadIdx.x) * 4;
    if (i + 3 < n) {
        #pragma unroll
        for (int c = 0; c < 4; c++) y[i + c] = (bf16_t)x[i + c];
    }
}

// ---------------------------------------------------------------------------
// Weight transpose + convert: W[K][N] fp32 -> Wt[N][K] bf16
// ---------------------------------------------------------------------------
__global__ __launch_bounds__(256) void wt_kernel(const float* __restrict__ W,
                                                 bf16_t* __restrict__ Wt,
                                                 int K, int N) {
    __shared__ float tile[32][33];
    const int tx = threadIdx.x & 31, ty = threadIdx.x >> 5;   // 32 x 8
    const int n0 = blockIdx.x * 32, k0 = blockIdx.y * 32;
    #pragma unroll
    for (int j = 0; j < 32; j += 8)
        tile[ty + j][tx] = W[(size_t)(k0 + ty + j) * N + n0 + tx];
    __syncthreads();
    #pragma unroll
    for (int j = 0; j < 32; j += 8)
        Wt[(size_t)(n0 + ty + j) * K + k0 + tx] = (bf16_t)tile[tx][ty + j];
}

// ---------------------------------------------------------------------------
// bf16 transpose: V[bh][S][64] -> Vt[bh][64][S]
// ---------------------------------------------------------------------------
__global__ __launch_bounds__(256) void vt_kernel(const bf16_t* __restrict__ V,
                                                 bf16_t* __restrict__ Vt) {
    __shared__ bf16_t tile[32][34];
    const int tx = threadIdx.x & 31, ty = threadIdx.x >> 5;
    const int s0 = blockIdx.x * 32, d0 = blockIdx.y * 32, bh = blockIdx.z;
    #pragma unroll
    for (int j = 0; j < 32; j += 8)
        tile[ty + j][tx] = V[((size_t)bh * SEQ + s0 + ty + j) * HEADD + d0 + tx];
    __syncthreads();
    #pragma unroll
    for (int j = 0; j < 32; j += 8)
        Vt[((size_t)bh * HEADD + d0 + ty + j) * SEQ + s0 + tx] = tile[tx][ty + j];
}

// ---------------------------------------------------------------------------
// Tiled bf16 GEMM with WMMA + double-buffered async global->LDS staging.
// C[M,N] = A[M,K] * Bt[N,K]^T + bias.  256 threads (8 waves), tile 128x128,
// K-step 64, ONE barrier per K-step (async pipeline covers the other hazard).
// MODE 0: store bf16 head-major (QKV)   MODE 1: gelu -> bf16   MODE 2: fp32
// ---------------------------------------------------------------------------
template <int MODE>
__global__ __launch_bounds__(256) void gemm_kernel(const bf16_t* __restrict__ A,
                                                   const bf16_t* __restrict__ Bt,
                                                   const float* __restrict__ bias,
                                                   void* __restrict__ outp,
                                                   int M, int N, int K) {
    constexpr int LD = 72;                 // padded stride (mult of 8 elems)
    constexpr int TILE = 128 * LD;         // elements per buffer
    __shared__ bf16_t sA[2 * TILE];
    __shared__ bf16_t sB[2 * TILE];
    const int tid  = threadIdx.x;
    const int lane = tid & 31, wid = tid >> 5;
    const int wm = wid >> 1, wn = wid & 1; // 4x2 wave grid over 128x128
    const int m0 = blockIdx.y * 128, n0 = blockIdx.x * 128;

    v8f zero = {};
    v8f acc[2][4];
    #pragma unroll
    for (int i = 0; i < 2; i++)
        #pragma unroll
        for (int j = 0; j < 4; j++) acc[i][j] = zero;

    // staging: thread -> (row, 64B half-row)
    const int srow = tid >> 1, shalf = tid & 1;
    const bf16_t* gA = A  + (size_t)(m0 + srow) * K + shalf * 32;
    const bf16_t* gB = Bt + (size_t)(n0 + srow) * K + shalf * 32;
    const uint32_t laA = lds_off_u32(sA + srow * LD + shalf * 32);
    const uint32_t laB = lds_off_u32(sB + srow * LD + shalf * 32);
    constexpr uint32_t BUFB = TILE * (uint32_t)sizeof(bf16_t);

    // prologue: stage tile 0 into buffer 0
    async_copy64(laA, (uint64_t)(uintptr_t)gA);
    async_copy64(laB, (uint64_t)(uintptr_t)gB);

    int buf = 0;
    for (int k = 0; k < K; k += 64) {
        wait_async0();
        __syncthreads();
        if (k + 64 < K) {                  // stage next tile into other buffer
            async_copy64(laA + (buf ^ 1) * BUFB, (uint64_t)(uintptr_t)(gA + k + 64));
            async_copy64(laB + (buf ^ 1) * BUFB, (uint64_t)(uintptr_t)(gB + k + 64));
        }
        if (k + 128 < K) {                 // deeper prefetch into L2/L0
            __builtin_prefetch(gA + k + 128, 0, 0);
            __builtin_prefetch(gB + k + 128, 0, 0);
        }
        const bf16_t* curA = sA + buf * TILE;
        const bf16_t* curB = sB + buf * TILE;
        #pragma unroll
        for (int kk = 0; kk < 64; kk += 32) {
            v16bf af[2], bfr[4];
            #pragma unroll
            for (int i = 0; i < 2; i++)
                af[i] = load_a_frag(curA + (wm * 32 + i * 16) * LD + kk, LD);
            #pragma unroll
            for (int j = 0; j < 4; j++)
                bfr[j] = load_b_frag(curB + (wn * 64 + j * 16) * LD + kk, LD);
            #pragma unroll
            for (int i = 0; i < 2; i++)
                #pragma unroll
                for (int j = 0; j < 4; j++)
                    acc[i][j] = wmma_bf16(af[i], bfr[j], acc[i][j]);
        }
        buf ^= 1;
    }

    // Epilogue. C layout: VGPR r, lane l -> row = r + 8*(l>=16), col = l&15.
    const int hi = lane >> 4, cn = lane & 15;
    #pragma unroll
    for (int i = 0; i < 2; i++) {
        #pragma unroll
        for (int j = 0; j < 4; j++) {
            const int col = n0 + wn * 64 + j * 16 + cn;
            const float bv = bias[col];
            #pragma unroll
            for (int r = 0; r < 8; r++) {
                const int row = m0 + wm * 32 + i * 16 + r + 8 * hi;
                float v = acc[i][j][r] + bv;
                if (MODE == 0) {
                    const int bb = row >> 11, s = row & (SEQ - 1);
                    const int h = col >> 6,  d = col & (HEADD - 1);
                    ((bf16_t*)outp)[(((size_t)(bb * NHEADS + h)) * SEQ + s) * HEADD + d] = (bf16_t)v;
                } else if (MODE == 1) {
                    float g = 0.5f * v * (1.0f + erff(v * 0.70710678118654752f));
                    g = fminf(fmaxf(g, -1e9f), 1e9f);
                    ((bf16_t*)outp)[(size_t)row * N + col] = (bf16_t)g;
                } else {
                    ((float*)outp)[(size_t)row * N + col] = v;
                }
            }
        }
    }
}

// ---------------------------------------------------------------------------
// Flash attention: one block (4 waves) per (64-query tile, head, batch).
// scores = Q K^T / 8, clamp, online softmax, O += P V.  HD = 64.
// ---------------------------------------------------------------------------
__global__ __launch_bounds__(128) void attn_kernel(const bf16_t* __restrict__ Q,
                                                   const bf16_t* __restrict__ Kb,
                                                   const bf16_t* __restrict__ Vt,
                                                   float* __restrict__ ctx) {
    constexpr int LP = 72;                 // P-tile LDS stride
    __shared__ bf16_t sP[64 * LP];
    const int tid = threadIdx.x, lane = tid & 31, w = tid >> 5;
    const int b = blockIdx.z, h = blockIdx.y;
    const int bh = b * NHEADS + h;
    const int q0 = blockIdx.x * 64;
    const bf16_t* qbase  = Q  + (size_t)bh * SEQ * HEADD;
    const bf16_t* kbase  = Kb + (size_t)bh * SEQ * HEADD;
    const bf16_t* vtbase = Vt + (size_t)bh * HEADD * SEQ;

    v8f zero = {};
    v8f o[4];
    #pragma unroll
    for (int j = 0; j < 4; j++) o[j] = zero;
    v8f mrun, lrun;
    #pragma unroll
    for (int r = 0; r < 8; r++) { mrun[r] = -3.0e38f; lrun[r] = 0.0f; }

    // Q fragments for this wave's 16 rows (reused all chunks)
    v16bf qf[2];
    #pragma unroll
    for (int kk = 0; kk < 2; kk++)
        qf[kk] = load_a_frag(qbase + (size_t)(q0 + w * 16) * HEADD + kk * 32, HEADD);

    for (int kc = 0; kc < SEQ; kc += 64) {
        // ---- scores: 16 rows x 64 keys ----
        v8f sf[4];
        #pragma unroll
        for (int j = 0; j < 4; j++) sf[j] = zero;
        #pragma unroll
        for (int kk = 0; kk < 2; kk++) {
            v16bf kf[4];
            #pragma unroll
            for (int j = 0; j < 4; j++)
                kf[j] = load_b_frag(kbase + (size_t)(kc + j * 16) * HEADD + kk * 32, HEADD);
            #pragma unroll
            for (int j = 0; j < 4; j++) sf[j] = wmma_bf16(qf[kk], kf[j], sf[j]);
        }
        #pragma unroll
        for (int j = 0; j < 4; j++)
            #pragma unroll
            for (int r = 0; r < 8; r++) {
                float v = sf[j][r] * 0.125f;           // 1/sqrt(64)
                sf[j][r] = fminf(fmaxf(v, -1e9f), 1e9f);
            }
        // ---- row max across 64 keys (lanes within 16-group share a row) ----
        v8f cmax;
        #pragma unroll
        for (int r = 0; r < 8; r++)
            cmax[r] = fmaxf(fmaxf(sf[0][r], sf[1][r]), fmaxf(sf[2][r], sf[3][r]));
        #pragma unroll
        for (int mask = 1; mask < 16; mask <<= 1)
            #pragma unroll
            for (int r = 0; r < 8; r++)
                cmax[r] = fmaxf(cmax[r], __shfl_xor(cmax[r], mask, 32));
        // ---- online softmax update ----
        v8f nm, scl;
        #pragma unroll
        for (int r = 0; r < 8; r++) {
            nm[r]  = fmaxf(mrun[r], cmax[r]);
            scl[r] = __expf(mrun[r] - nm[r]);
            mrun[r] = nm[r];
        }
        #pragma unroll
        for (int j = 0; j < 4; j++)
            #pragma unroll
            for (int r = 0; r < 8; r++) o[j][r] *= scl[r];
        v8f csum;
        #pragma unroll
        for (int r = 0; r < 8; r++) csum[r] = 0.0f;
        #pragma unroll
        for (int j = 0; j < 4; j++)
            #pragma unroll
            for (int r = 0; r < 8; r++) {
                float p = __expf(sf[j][r] - nm[r]);
                sf[j][r] = p;
                csum[r] += p;
            }
        #pragma unroll
        for (int mask = 1; mask < 16; mask <<= 1)
            #pragma unroll
            for (int r = 0; r < 8; r++) csum[r] += __shfl_xor(csum[r], mask, 32);
        #pragma unroll
        for (int r = 0; r < 8; r++) lrun[r] = lrun[r] * scl[r] + csum[r];
        // ---- P -> LDS (re-fragment as A) ----
        {
            const int hi = lane >> 4, cn = lane & 15;
            #pragma unroll
            for (int j = 0; j < 4; j++)
                #pragma unroll
                for (int r = 0; r < 8; r++)
                    sP[(w * 16 + r + 8 * hi) * LP + j * 16 + cn] = (bf16_t)sf[j][r];
        }
        __syncthreads();
        // ---- O += P * V ----
        #pragma unroll
        for (int kk = 0; kk < 2; kk++) {
            v16bf pf = load_a_frag(sP + (w * 16) * LP + kk * 32, LP);
            v16bf vf[4];
            #pragma unroll
            for (int j = 0; j < 4; j++)
                vf[j] = load_b_frag(vtbase + (size_t)(j * 16) * SEQ + kc + kk * 32, SEQ);
            #pragma unroll
            for (int j = 0; j < 4; j++) o[j] = wmma_bf16(pf, vf[j], o[j]);
        }
        __syncthreads();
    }

    // ---- normalize and store ctx fp32 [B][S][H] ----
    const int hi = lane >> 4, cn = lane & 15;
    v8f inv;
    #pragma unroll
    for (int r = 0; r < 8; r++) inv[r] = 1.0f / lrun[r];
    #pragma unroll
    for (int j = 0; j < 4; j++)
        #pragma unroll
        for (int r = 0; r < 8; r++) {
            const int row = q0 + w * 16 + r + 8 * hi;
            const int d = j * 16 + cn;
            ctx[((size_t)b * SEQ + row) * HDIM + h * HEADD + d] = o[j][r] * inv[r];
        }
}

// ---------------------------------------------------------------------------
// LayerNorm(a + bsrc) * g + be  -> outf (fp32) and optionally outb (bf16)
// One wave per 1024-wide row; 8 rows per block.
// ---------------------------------------------------------------------------
__global__ __launch_bounds__(256) void ln_kernel(const float* __restrict__ a,
                                                 const float* __restrict__ bsrc,
                                                 const float* __restrict__ g,
                                                 const float* __restrict__ be,
                                                 float* __restrict__ outf,
                                                 bf16_t* __restrict__ outb) {
    const int lane = threadIdx.x & 31, wid = threadIdx.x >> 5;
    const int row = blockIdx.x * 8 + wid;
    const float* pa = a    + (size_t)row * HDIM;
    const float* pb = bsrc + (size_t)row * HDIM;
    float vals[32];
    float sum = 0.0f;
    #pragma unroll
    for (int i = 0; i < 32; i++) {
        const int c = lane + i * 32;
        vals[i] = pa[c] + pb[c];
        sum += vals[i];
    }
    #pragma unroll
    for (int mask = 16; mask >= 1; mask >>= 1) sum += __shfl_xor(sum, mask, 32);
    const float mu = sum * (1.0f / HDIM);
    float var = 0.0f;
    #pragma unroll
    for (int i = 0; i < 32; i++) { float d = vals[i] - mu; var += d * d; }
    #pragma unroll
    for (int mask = 16; mask >= 1; mask >>= 1) var += __shfl_xor(var, mask, 32);
    var *= (1.0f / HDIM);
    const float rs = rsqrtf(var + 1e-12f);
    #pragma unroll
    for (int i = 0; i < 32; i++) {
        const int c = lane + i * 32;
        const float y = (vals[i] - mu) * rs * g[c] + be[c];
        outf[(size_t)row * HDIM + c] = y;
        if (outb) outb[(size_t)row * HDIM + c] = (bf16_t)y;
    }
}

// ---------------------------------------------------------------------------
// Launch
// ---------------------------------------------------------------------------
extern "C" void kernel_launch(void* const* d_in, const int* in_sizes, int n_in,
                              void* d_out, int out_size, void* d_ws, size_t ws_size,
                              hipStream_t stream) {
    const float* hs   = (const float*)d_in[0];
    const float* Wq   = (const float*)d_in[1];
    const float* bq   = (const float*)d_in[2];
    const float* Wk   = (const float*)d_in[3];
    const float* bk   = (const float*)d_in[4];
    const float* Wv   = (const float*)d_in[5];
    const float* bv   = (const float*)d_in[6];
    const float* ln1g = (const float*)d_in[7];
    const float* ln1b = (const float*)d_in[8];
    const float* W1   = (const float*)d_in[9];
    const float* b1   = (const float*)d_in[10];
    const float* W2   = (const float*)d_in[11];
    const float* b2   = (const float*)d_in[12];
    const float* ln2g = (const float*)d_in[13];
    const float* ln2b = (const float*)d_in[14];
    float* out = (float*)d_out;

    const size_t MTOK = (size_t)BATCH * SEQ;               // 8192 rows
    char* base = (char*)d_ws;
    size_t off = 0;
    auto alloc = [&](size_t bytes) -> void* {
        off = (off + 255) & ~(size_t)255;
        void* p = base + off;
        off += bytes;
        return p;
    };

    // NOTE: hsb/qb/kb/vb allocated back-to-back; their combined region is
    // reused as `inter` (all four are dead before FFN1 writes it).
    bf16_t* hsb = (bf16_t*)alloc(MTOK * HDIM * 2);                 // 16 MB
    bf16_t* qb  = (bf16_t*)alloc(MTOK * HDIM * 2);
    bf16_t* kb  = (bf16_t*)alloc(MTOK * HDIM * 2);
    bf16_t* vb  = (bf16_t*)alloc(MTOK * HDIM * 2);
    bf16_t* inter = hsb;                                           // 64 MB alias
    bf16_t* vtb = (bf16_t*)alloc(MTOK * HDIM * 2);
    bf16_t* wqt = (bf16_t*)alloc((size_t)HDIM * HDIM * 2);
    bf16_t* wkt = (bf16_t*)alloc((size_t)HDIM * HDIM * 2);
    bf16_t* wvt = (bf16_t*)alloc((size_t)HDIM * HDIM * 2);
    bf16_t* w1t = (bf16_t*)alloc((size_t)HDIM * FFDIM * 2);
    bf16_t* w2t = (bf16_t*)alloc((size_t)FFDIM * HDIM * 2);
    float*  ctx = (float*)alloc(MTOK * HDIM * 4);                  // reused as ffn out
    float*  x   = (float*)alloc(MTOK * HDIM * 4);
    bf16_t* xb  = (bf16_t*)alloc(MTOK * HDIM * 2);
    (void)ws_size; (void)in_sizes; (void)n_in; (void)out_size;

    // 1. converts / transposes
    cvt_kernel<<<(MTOK * HDIM) / 1024, 256, 0, stream>>>(hs, hsb, MTOK * HDIM);
    wt_kernel<<<dim3(HDIM / 32, HDIM / 32), 256, 0, stream>>>(Wq, wqt, HDIM, HDIM);
    wt_kernel<<<dim3(HDIM / 32, HDIM / 32), 256, 0, stream>>>(Wk, wkt, HDIM, HDIM);
    wt_kernel<<<dim3(HDIM / 32, HDIM / 32), 256, 0, stream>>>(Wv, wvt, HDIM, HDIM);
    wt_kernel<<<dim3(FFDIM / 32, HDIM / 32), 256, 0, stream>>>(W1, w1t, HDIM, FFDIM);
    wt_kernel<<<dim3(HDIM / 32, FFDIM / 32), 256, 0, stream>>>(W2, w2t, FFDIM, HDIM);

    // 2. QKV projections -> head-major bf16
    gemm_kernel<0><<<dim3(HDIM / 128, MTOK / 128), 256, 0, stream>>>(hsb, wqt, bq, qb, MTOK, HDIM, HDIM);
    gemm_kernel<0><<<dim3(HDIM / 128, MTOK / 128), 256, 0, stream>>>(hsb, wkt, bk, kb, MTOK, HDIM, HDIM);
    gemm_kernel<0><<<dim3(HDIM / 128, MTOK / 128), 256, 0, stream>>>(hsb, wvt, bv, vb, MTOK, HDIM, HDIM);
    vt_kernel<<<dim3(SEQ / 32, HEADD / 32, BATCH * NHEADS), 256, 0, stream>>>(vb, vtb);

    // 3. attention
    attn_kernel<<<dim3(SEQ / 64, NHEADS, BATCH), 128, 0, stream>>>(qb, kb, vtb, ctx);

    // 4. residual + LN1 -> x (fp32) + xb (bf16)
    ln_kernel<<<MTOK / 8, 256, 0, stream>>>(hs, ctx, ln1g, ln1b, x, xb);

    // 5. FFN
    gemm_kernel<1><<<dim3(FFDIM / 128, MTOK / 128), 256, 0, stream>>>(xb, w1t, b1, inter, MTOK, FFDIM, HDIM);
    gemm_kernel<2><<<dim3(HDIM / 128, MTOK / 128), 256, 0, stream>>>(inter, w2t, b2, ctx, MTOK, HDIM, FFDIM);

    // 6. residual + LN2 -> out
    ln_kernel<<<MTOK / 8, 256, 0, stream>>>(ctx, x, ln2g, ln2b, out, nullptr);
}
